// vipUnidirectional_54915451846850
// MI455X (gfx1250) — matrix operand order repaired and checked by reading, not compiled, not run on hardware
//
#include <hip/hip_runtime.h>
#include <math.h>

typedef __bf16        v16bf __attribute__((ext_vector_type(16)));
typedef float         v8f   __attribute__((ext_vector_type(8)));
typedef unsigned int  u32x8 __attribute__((ext_vector_type(8)));
typedef unsigned int  u32x4 __attribute__((ext_vector_type(4)));

constexpr int   BB   = 64;     // batch
constexpr int   TT   = 32;     // words
constexpr int   CC   = 256;    // NEF
constexpr int   SPIX = 289;    // 17*17 regions
constexpr int   SST  = 324;    // ctxp row stride (u32 words), %4==0 -> 16B-aligned rows
constexpr int   SROWS= 304;    // 19*16 padded s rows for score matrix
constexpr int   KPAD = 320;    // 10*32 padded K for wC matmul
constexpr int   KQ   = 16384;  // queue length
constexpr float G1f = 4.0f, G2f = 5.0f, G3f = 10.0f;
constexpr float EPSf = 1e-8f;
constexpr float INV_T = 20.0f; // 1/0.05

// LDS byte layout for k_attn (all operands stored as k-pair-interleaved u32)
constexpr int CTXP_B = (CC / 2) * SST * 4;    // 165888  ctxp[c/2][s]
constexpr int LQP_B  = (CC / 2) * TT * 4;     // 16384   lqp[c/2][t]
constexpr int LS_B   = SROWS * TT * 4;        // 38912   lS[s][t] f32
constexpr int LATP_B = (KPAD / 2) * TT * 4;   // 20480   lattnp[s/2][t]
constexpr int SMEM_ATTN = CTXP_B + LQP_B + LS_B + LATP_B + 3 * TT * 4; // 242048

// workspace float offsets
constexpr int OFF_SIM = 0;                     // simMat[64*64]  (S rows = j)
constexpr int OFF_CN  = 4096;                  // cnN, rnN, canN, ranN (4 x 64*256)
constexpr int OFF_LSE = OFF_CN + 4 * BB * CC;  // lse_it[64], lse_ti[64]
constexpr int OFF_POS = OFF_LSE + 2 * BB;      // pos_it[64], pos_ti[64]

static __device__ __forceinline__ unsigned int f2bf(float f) {
  unsigned int u = __float_as_uint(f);
  u += 0x7FFFu + ((u >> 16) & 1u);            // round-to-nearest-even
  return u >> 16;
}
static __device__ __forceinline__ float us2f(unsigned int u16) {
  return __uint_as_float(u16 << 16);
}

// ---------------------------------------------------------------------------
// Kernel 0: L2-normalize the four code matrices [64,256] -> ws
// ---------------------------------------------------------------------------
__global__ void __launch_bounds__(256) k_normalize(const float* __restrict__ c0,
                                                   const float* __restrict__ c1,
                                                   const float* __restrict__ c2,
                                                   const float* __restrict__ c3,
                                                   float* __restrict__ outb) {
  __shared__ float red[8];
  const int r = blockIdx.x, tid = threadIdx.x;
  const float* in[4] = {c0, c1, c2, c3};
#pragma unroll
  for (int q = 0; q < 4; ++q) {
    float v = in[q][r * CC + tid];
    float s = v * v;
    for (int off = 16; off; off >>= 1) s += __shfl_xor(s, off, 32);
    if ((tid & 31) == 0) red[tid >> 5] = s;
    __syncthreads();
    float tot = red[0];
#pragma unroll
    for (int k = 1; k < 8; ++k) tot += red[k];
    outb[q * (BB * CC) + r * CC + tid] = v / fmaxf(sqrtf(tot), EPSf);
    __syncthreads();
  }
}

// ---------------------------------------------------------------------------
// Kernel 1: word-region attention -> simMat[j*64+i] = G3 * sim(i,j)
// One WG per (image j, block of 16 texts). ctx[j] staged once in ~242KB LDS
// as bf16 pairs; both GEMMs on v_wmma_f32_16x16x32_bf16. B-fragments are
// hoisted into registers (t-tile fixed per wave); wC A-fragments use b128.
// ---------------------------------------------------------------------------
__global__ void __launch_bounds__(256) k_attn(const float* __restrict__ img,
                                              const float* __restrict__ words,
                                              float* __restrict__ simMat) {
  extern __shared__ unsigned char smem[];
  unsigned int* ctxp   = (unsigned int*)smem;                                // [c/2][s]
  unsigned int* lqp    = (unsigned int*)(smem + CTXP_B);                     // [c/2][t]
  float*        lS     = (float*)(smem + CTXP_B + LQP_B);                    // [s][t]
  unsigned int* lattnp = (unsigned int*)(smem + CTXP_B + LQP_B + LS_B);      // [s/2][t]
  float*        lqn    = (float*)(smem + CTXP_B + LQP_B + LS_B + LATP_B);
  float*        lnum   = lqn + TT;
  float*        lwn    = lnum + TT;

  const int tid  = threadIdx.x;
  const int lane = tid & 31;
  const int wid  = tid >> 5;
  const int m16  = lane & 15;
  const int hig  = lane >> 4;
  const int j    = blockIdx.x;
  const int ibl  = blockIdx.y;
  const int ttile = wid & 1;                 // fixed N-tile per wave
  const int nB    = ttile * 16 + m16;        // this lane's matrix column (t)

  // Phase A: stage ctx[j] (256 x 289 f32) as bf16 c-pairs, zero-pad s in [289,324)
  for (int w = wid; w < CC / 2; w += 8) {
    const float* s0p = img + ((size_t)j * CC + 2 * w) * SPIX;
    const float* s1p = s0p + SPIX;
    for (int s = lane; s < SST; s += 32) {
      unsigned int pk = 0u;
      if (s < SPIX) pk = f2bf(s0p[s]) | (f2bf(s1p[s]) << 16);
      ctxp[w * SST + s] = pk;
    }
  }
  __syncthreads();

  for (int ii = 0; ii < 16; ++ii) {
    const int i = ibl * 16 + ii;
    const float* wsrc = words + (size_t)i * TT * CC;   // [t][c]

    // B1: stage q[i] -> lqp[c/2][t] (bf16 pairs); word norms; zero reductions
    for (int idx = tid; idx < TT * (CC / 2); idx += 256) {
      const int t = idx >> 7, w = idx & (CC / 2 - 1);
      const float* p = wsrc + t * CC + 2 * w;
      lqp[w * TT + t] = f2bf(p[0]) | (f2bf(p[1]) << 16);
    }
    if (tid < TT) {
      float s = 0.f;
      for (int c = 0; c < CC; ++c) { float v = wsrc[tid * CC + c]; s += v * v; }
      lqn[tid]  = sqrtf(s);
      lnum[tid] = 0.f;
      lwn[tid]  = 0.f;
    }
    __syncthreads();

    // B2: scores S[s][t] = sum_c ctx[c][s] * q[c][t]   (19 s-tiles, K=256)
    {
      // Hoist the 8 q B-fragments for this wave's t-tile into registers.
      u32x8 bfr[8];
#pragma unroll
      for (int kc = 0; kc < 8; ++kc) {
        const unsigned int* bb = lqp + (kc * 16 + 4 * hig) * TT + nB;
#pragma unroll
        for (int v = 0; v < 8; ++v) bfr[kc][v] = bb[(v + (v & 4)) * TT];
      }
      for (int tile = wid; tile < 38; tile += 8) {
        const int stile = tile >> 1;
        const int sA = stile * 16 + m16;      // A row (s)
        v8f acc = {};
#pragma unroll
        for (int kc = 0; kc < 8; ++kc) {
          const unsigned int* ab = ctxp + (kc * 16 + 4 * hig) * SST + sA;
          u32x8 aw;
          aw[0] = ab[0 * SST];  aw[1] = ab[1 * SST];
          aw[2] = ab[2 * SST];  aw[3] = ab[3 * SST];
          aw[4] = ab[8 * SST];  aw[5] = ab[9 * SST];
          aw[6] = ab[10 * SST]; aw[7] = ab[11 * SST];
          const v16bf a = __builtin_bit_cast(v16bf, aw);
          const v16bf b = __builtin_bit_cast(v16bf, bfr[kc]);
          acc = __builtin_amdgcn_wmma_f32_16x16x32_bf16(false, a, false, b,
                                                        (short)0, acc, false, false);
        }
#pragma unroll
        for (int r = 0; r < 8; ++r)
          lS[(stile * 16 + 8 * hig + r) * TT + ttile * 16 + m16] = acc[r];
      }
    }
    __syncthreads();

    // B3a: softmax over words t (rows of lS, in place)
    for (int s = tid; s < SPIX; s += 256) {
      float* row = lS + s * TT;
      float mx = row[0];
#pragma unroll
      for (int t = 1; t < TT; ++t) mx = fmaxf(mx, row[t]);
      float sum = 0.f;
#pragma unroll
      for (int t = 0; t < TT; ++t) sum += __expf(row[t] - mx);
      const float inv = 1.0f / sum;
#pragma unroll
      for (int t = 0; t < TT; ++t) row[t] = __expf(row[t] - mx) * inv;
    }
    __syncthreads();

    // B3b: softmax over regions s (cols) of G1*attn1 -> packed bf16 attn pairs
    if (tid < TT) {
      const int t = tid;
      float mx = -1e30f;
      for (int s = 0; s < SPIX; ++s) mx = fmaxf(mx, lS[s * TT + t]);
      mx *= G1f;
      float sum = 0.f;
      for (int s = 0; s < SPIX; ++s) sum += __expf(G1f * lS[s * TT + t] - mx);
      const float inv = 1.0f / sum;
      for (int s2 = 0; s2 < KPAD / 2; ++s2) {
        const int s0 = 2 * s2, s1 = s0 + 1;
        unsigned int lo = (s0 < SPIX) ? f2bf(__expf(G1f * lS[s0 * TT + t] - mx) * inv) : 0u;
        unsigned int hi = (s1 < SPIX) ? f2bf(__expf(G1f * lS[s1 * TT + t] - mx) * inv) : 0u;
        lattnp[s2 * TT + t] = lo | (hi << 16);
      }
    }
    __syncthreads();

    // B4: wC[c][t] = sum_s ctx[c][s]*attn[s][t]; fuse num/wn reductions
    {
      // Hoist the 10 attn B-fragments for this wave's t-tile into registers.
      u32x8 bfr[10];
#pragma unroll
      for (int ks = 0; ks < 10; ++ks) {
        const unsigned int* bb = lattnp + (ks * 16 + 4 * hig) * TT + nB;
#pragma unroll
        for (int v = 0; v < 8; ++v) bfr[ks][v] = bb[(v + (v & 4)) * TT];
      }
      float num_p = 0.f, wn_p = 0.f;
      for (int tile = wid; tile < 32; tile += 8) {
        const int ctile = tile >> 1;
        const int cA = ctile * 16 + m16;
        const unsigned int psel = (cA & 1) ? 0x07060302u : 0x05040100u;
        v8f acc = {};
#pragma unroll
        for (int ks = 0; ks < 10; ++ks) {
          // A row base: 16B-aligned (SST%4==0, kb%4==0, 8*hig%4==0)
          const u32x4* ab4 = (const u32x4*)(ctxp + (cA >> 1) * SST + ks * 32 + 8 * hig);
          const u32x4 q0 = ab4[0];   // words 0..3
          const u32x4 q1 = ab4[1];   // words 4..7
          const u32x4 q2 = ab4[4];   // words 16..19
          const u32x4 q3 = ab4[5];   // words 20..23
          u32x8 aw;
          aw[0] = __builtin_amdgcn_perm(q0[1], q0[0], psel);
          aw[1] = __builtin_amdgcn_perm(q0[3], q0[2], psel);
          aw[2] = __builtin_amdgcn_perm(q1[1], q1[0], psel);
          aw[3] = __builtin_amdgcn_perm(q1[3], q1[2], psel);
          aw[4] = __builtin_amdgcn_perm(q2[1], q2[0], psel);
          aw[5] = __builtin_amdgcn_perm(q2[3], q2[2], psel);
          aw[6] = __builtin_amdgcn_perm(q3[1], q3[0], psel);
          aw[7] = __builtin_amdgcn_perm(q3[3], q3[2], psel);
          const v16bf a = __builtin_bit_cast(v16bf, aw);
          const v16bf b = __builtin_bit_cast(v16bf, bfr[ks]);
          acc = __builtin_amdgcn_wmma_f32_16x16x32_bf16(false, a, false, b,
                                                        (short)0, acc, false, false);
        }
        const int t = nB;
#pragma unroll
        for (int r = 0; r < 8; ++r) {
          const int c = ctile * 16 + 8 * hig + r;
          const unsigned int qw = lqp[(c >> 1) * TT + t];
          const float qv = us2f((c & 1) ? (qw >> 16) : (qw & 0xFFFFu));
          const float w = acc[r];
          num_p += w * qv;
          wn_p  += w * w;
        }
      }
      atomicAdd(&lnum[nB], num_p);
      atomicAdd(&lwn[nB], wn_p);
    }
    __syncthreads();

    // B5: cosine + logsumexp over t -> simMat[j][i] (already * G3)
    if (tid < TT) {
      const int t = tid;
      const float denom = fmaxf(lqn[t] * sqrtf(lwn[t]), EPSf);
      const float val = G2f * (lnum[t] / denom);
      float mx = val;
      for (int off = 16; off; off >>= 1) mx = fmaxf(mx, __shfl_xor(mx, off, 32));
      float e = __expf(val - mx);
      for (int off = 16; off; off >>= 1) e += __shfl_xor(e, off, 32);
      if (t == 0) simMat[j * BB + i] = G3f * (mx + __logf(e));
    }
    __syncthreads();
  }
}

// ---------------------------------------------------------------------------
// Kernel 2: MoCo logits: per row i, lse over [pos, x_i . Q]/T  (streaming,
// memory-bound: 32MB of queue reads once). blockIdx.y: 0=it, 1=ti.
// ---------------------------------------------------------------------------
__global__ void __launch_bounds__(256) k_moco(const float* __restrict__ base,
                                              const float* __restrict__ queue,
                                              const float* __restrict__ queue_im,
                                              float* __restrict__ lse_out,
                                              float* __restrict__ pos_out) {
  __shared__ float lx[CC];
  __shared__ float redm[8], reds[8], redp[8];
  __shared__ float sh_pos;
  const int i = blockIdx.x, which = blockIdx.y, tid = threadIdx.x;
  const float* x   = base + (which ? (BB * CC)     : 0)             + i * CC; // rn : cn
  const float* aug = base + (which ? (2 * BB * CC) : (3 * BB * CC)) + i * CC; // can : ran
  const float* Q   = which ? queue_im : queue;

  const float xv = x[tid];
  lx[tid] = xv;
  float p = xv * aug[tid];
  for (int off = 16; off; off >>= 1) p += __shfl_xor(p, off, 32);
  if ((tid & 31) == 0) redp[tid >> 5] = p;
  __syncthreads();
  if (tid == 0) {
    float t = 0.f;
    for (int k = 0; k < 8; ++k) t += redp[k];
    sh_pos = t * INV_T;
  }
  __syncthreads();

  float mx = -1e30f, sm = 0.f;
  for (int k = tid; k < KQ; k += 256) {
    float d = 0.f;
    for (int c = 0; c < CC; ++c) d += lx[c] * Q[(size_t)c * KQ + k];
    d *= INV_T;
    if (d > mx) { sm = sm * __expf(mx - d) + 1.0f; mx = d; }
    else        { sm += __expf(d - mx); }
  }
  for (int off = 16; off; off >>= 1) {
    const float om = __shfl_xor(mx, off, 32), os = __shfl_xor(sm, off, 32);
    const float nm = fmaxf(mx, om);
    sm = sm * __expf(mx - nm) + os * __expf(om - nm);
    mx = nm;
  }
  if ((tid & 31) == 0) { redm[tid >> 5] = mx; reds[tid >> 5] = sm; }
  __syncthreads();
  if (tid == 0) {
    float M = redm[0], S = reds[0];
    for (int k = 1; k < 8; ++k) {
      const float nm = fmaxf(M, redm[k]);
      S = S * __expf(M - nm) + reds[k] * __expf(redm[k] - nm);
      M = nm;
    }
    const float pv = sh_pos;                  // positive logit (index 0)
    const float nm = fmaxf(M, pv);
    S = S * __expf(M - nm) + __expf(pv - nm);
    M = nm;
    lse_out[which * BB + i] = M + __logf(S);
    pos_out[which * BB + i] = pv;
  }
}

// ---------------------------------------------------------------------------
// Kernel 3: assemble the 7 losses (single block).
// ---------------------------------------------------------------------------
__global__ void __launch_bounds__(256) k_final(const float* __restrict__ ws,
                                               const int* __restrict__ labels,
                                               float* __restrict__ out) {
  __shared__ float s0sh[BB * BB], s2sh[BB * BB], smsh[BB * BB];
  __shared__ float accsh[2];
  const int tid = threadIdx.x;
  const float* cn  = ws + OFF_CN;
  const float* rn  = cn + BB * CC;
  const float* can = rn + BB * CC;
  const float* ran = can + BB * CC;
  const float* lse = ws + OFF_LSE;
  const float* pos = ws + OFF_POS;
  (void)can;

  for (int idx = tid; idx < BB * BB; idx += 256) {
    const int a = idx >> 6, b = idx & 63;
    float d0 = 0.f, d2 = 0.f;
    for (int c = 0; c < CC; ++c) {
      d0 += cn[a * CC + c] * rn[b * CC + c];
      d2 += rn[a * CC + c] * ran[b * CC + c];
    }
    s0sh[idx] = G3f * d0;
    s2sh[idx] = G3f * d2;
    smsh[idx] = ws[OFF_SIM + idx];
  }
  __syncthreads();

  auto ce = [&](const float* M, bool transposed) -> float {
    float partial = 0.f;
    if (tid < BB) {
      const int r = tid;
      const int lr = labels[r];
      float mx = -1e30f;
      for (int k = 0; k < BB; ++k) {
        const float v = transposed ? M[k * BB + r] : M[r * BB + k];
        mx = fmaxf(mx, v);
      }
      float s = 0.f;
      for (int k = 0; k < BB; ++k) {
        const float v = transposed ? M[k * BB + r] : M[r * BB + k];
        s += __expf(v - mx);
      }
      const float posv = transposed ? M[lr * BB + r] : M[r * BB + lr];
      partial = (mx + __logf(s)) - posv;
    }
    for (int off = 16; off; off >>= 1) partial += __shfl_xor(partial, off, 32);
    __syncthreads();
    if (tid == 0)  accsh[0] = partial;
    if (tid == 32) accsh[1] = partial;
    __syncthreads();
    const float res = (accsh[0] + accsh[1]) * (1.0f / 64.0f);
    __syncthreads();
    return res;
  };

  const float w0 = ce(smsh, false);
  const float w1 = ce(smsh, true);
  const float l0 = ce(s0sh, false);
  const float l1 = ce(s0sh, true);
  const float l2 = ce(s2sh, false);
  const float l3 = ce(s2sh, true);

  float pm = 0.f;
  if (tid < BB)
    pm = (lse[tid] - pos[tid]) + (lse[BB + tid] - pos[BB + tid]);
  for (int off = 16; off; off >>= 1) pm += __shfl_xor(pm, off, 32);
  __syncthreads();
  if (tid == 0)  accsh[0] = pm;
  if (tid == 32) accsh[1] = pm;
  __syncthreads();
  const float m_loss = (accsh[0] + accsh[1]) * (1.0f / 128.0f);

  if (tid == 0) {
    out[0] = w0; out[1] = w1; out[2] = l0; out[3] = l1;
    out[4] = m_loss; out[5] = l2; out[6] = l3;
  }
}

// ---------------------------------------------------------------------------
extern "C" void kernel_launch(void* const* d_in, const int* in_sizes, int n_in,
                              void* d_out, int out_size, void* d_ws, size_t ws_size,
                              hipStream_t stream) {
  const float* cnn = (const float*)d_in[0];
  const float* rnn = (const float*)d_in[1];
  const float* img = (const float*)d_in[2];
  const float* wrd = (const float*)d_in[3];
  const float* cna = (const float*)d_in[4];
  const float* rna = (const float*)d_in[5];
  const float* qz  = (const float*)d_in[6];
  const float* qzi = (const float*)d_in[7];
  const int*   lab = (const int*)d_in[9];
  float* ws  = (float*)d_ws;
  float* out = (float*)d_out;

  k_normalize<<<BB, 256, 0, stream>>>(cnn, rnn, cna, rna, ws + OFF_CN);
  k_attn<<<dim3(BB, 4), 256, SMEM_ATTN, stream>>>(img, wrd, ws + OFF_SIM);
  k_moco<<<dim3(BB, 2), 256, 0, stream>>>(ws + OFF_CN, qz, qzi,
                                          ws + OFF_LSE, ws + OFF_POS);
  k_final<<<1, 256, 0, stream>>>(ws, lab, out);
}
